// FedformerLayer_63900523430354
// MI455X (gfx1250) — matrix operand order repaired and checked by reading, not compiled, not run on hardware
//
#include <hip/hip_runtime.h>

typedef __attribute__((ext_vector_type(16))) __bf16 bf16x16;
typedef __attribute__((ext_vector_type(8)))  __bf16 bf16x8;
typedef __attribute__((ext_vector_type(8)))  float  f32x8;

#define S_LEN  4096
#define H_DIM  1024
#define KDIM   1024
#define NH     16
#define DK     64
#define NMODES 4
#define BATCH  4
#define TWO_PI 6.28318530717958647692f

// ---------------- K0: twiddle table tw[m][s] = (cos, sin) of 2*pi*index[m]*s/S ----
__global__ void twiddle_kernel(const int* __restrict__ index, float* __restrict__ tw) {
  int idx = blockIdx.x * 256 + threadIdx.x;      // over NMODES*S
  int m = idx >> 12, s = idx & (S_LEN - 1);
  int f = index[m];
  int r = (f * s) & (S_LEN - 1);                 // exact phase reduction (S = 2^12)
  float sn, cs;
  __sincosf((float)r * (TWO_PI / S_LEN), &sn, &cs);
  tw[idx * 2] = cs;
  tw[idx * 2 + 1] = sn;
}

// ---------------- K1: q = x @ Wq^T + bq  via split-bf16 WMMA ----------------------
// Macro tile 64(M) x 128(N), K-step 32. 8 waves (2x4), wave tile 32x32.
__global__ __launch_bounds__(256) void qproj_kernel(
    const float* __restrict__ x, const float* __restrict__ Wq,
    const float* __restrict__ bq, float* __restrict__ q) {
  __shared__ __align__(16) __bf16 Ah[64 * 40];
  __shared__ __align__(16) __bf16 Al[64 * 40];
  __shared__ __align__(16) __bf16 Bh[128 * 40];
  __shared__ __align__(16) __bf16 Bl[128 * 40];

  const int tid = threadIdx.x;
  const int m0 = blockIdx.x * 64;
  const int n0 = blockIdx.y * 128;
  const int wid = tid >> 5;
  const int lane = tid & 31;
  const int laneRow = lane & 15;
  const int laneHalf = lane >> 4;
  const int wm = wid >> 2;   // 0..1 (M)
  const int wn = wid & 3;    // 0..3 (N)

  f32x8 acc[2][2] = {};

  for (int k0 = 0; k0 < KDIM; k0 += 32) {
    __syncthreads();
    // ---- stage A tile (64 rows x 32 k), split into hi/lo bf16 ----
#pragma unroll
    for (int it = 0; it < 4; ++it) {
      int idx = tid + it * 256;
      int row = idx >> 4, kp = idx & 15;
      float2 v = reinterpret_cast<const float2*>(x + (size_t)(m0 + row) * KDIM + k0)[kp];
      __bf16 h0 = (__bf16)v.x; __bf16 l0 = (__bf16)(v.x - (float)h0);
      __bf16 h1 = (__bf16)v.y; __bf16 l1 = (__bf16)(v.y - (float)h1);
      Ah[row * 40 + 2 * kp] = h0; Ah[row * 40 + 2 * kp + 1] = h1;
      Al[row * 40 + 2 * kp] = l0; Al[row * 40 + 2 * kp + 1] = l1;
    }
    // ---- stage B tile (128 n x 32 k): Bt[n][k] = Wq[n][k] (B = Wq^T is KxN) ----
#pragma unroll
    for (int it = 0; it < 8; ++it) {
      int idx = tid + it * 256;
      int row = idx >> 4, kp = idx & 15;
      float2 v = reinterpret_cast<const float2*>(Wq + (size_t)(n0 + row) * KDIM + k0)[kp];
      __bf16 h0 = (__bf16)v.x; __bf16 l0 = (__bf16)(v.x - (float)h0);
      __bf16 h1 = (__bf16)v.y; __bf16 l1 = (__bf16)(v.y - (float)h1);
      Bh[row * 40 + 2 * kp] = h0; Bh[row * 40 + 2 * kp + 1] = h1;
      Bl[row * 40 + 2 * kp] = l0; Bl[row * 40 + 2 * kp + 1] = l1;
    }
    __syncthreads();

    // ---- fragments (ISA 16-bit A 16x32 / B 32x16 lane layouts) ----
    bf16x16 aH[2], aL[2], bH[2], bL[2];
#pragma unroll
    for (int mi = 0; mi < 2; ++mi) {
      int r = wm * 32 + mi * 16 + laneRow;
      int kb = laneHalf * 8;                       // lanes<16: K 0..7,16..23 ; lanes>=16: +8
      bf16x8 p0 = *reinterpret_cast<const bf16x8*>(&Ah[r * 40 + kb]);
      bf16x8 p1 = *reinterpret_cast<const bf16x8*>(&Ah[r * 40 + kb + 16]);
      aH[mi] = __builtin_shufflevector(p0, p1, 0,1,2,3,4,5,6,7,8,9,10,11,12,13,14,15);
      bf16x8 q0 = *reinterpret_cast<const bf16x8*>(&Al[r * 40 + kb]);
      bf16x8 q1 = *reinterpret_cast<const bf16x8*>(&Al[r * 40 + kb + 16]);
      aL[mi] = __builtin_shufflevector(q0, q1, 0,1,2,3,4,5,6,7,8,9,10,11,12,13,14,15);
    }
#pragma unroll
    for (int ni = 0; ni < 2; ++ni) {
      int n = wn * 32 + ni * 16 + laneRow;
      int kb = laneHalf * 8;                       // lanes<16: K 0..7,16..23 ; lanes>=16: +8
      bf16x8 p0 = *reinterpret_cast<const bf16x8*>(&Bh[n * 40 + kb]);
      bf16x8 p1 = *reinterpret_cast<const bf16x8*>(&Bh[n * 40 + kb + 16]);
      bH[ni] = __builtin_shufflevector(p0, p1, 0,1,2,3,4,5,6,7,8,9,10,11,12,13,14,15);
      bf16x8 q0 = *reinterpret_cast<const bf16x8*>(&Bl[n * 40 + kb]);
      bf16x8 q1 = *reinterpret_cast<const bf16x8*>(&Bl[n * 40 + kb + 16]);
      bL[ni] = __builtin_shufflevector(q0, q1, 0,1,2,3,4,5,6,7,8,9,10,11,12,13,14,15);
    }
    // ---- split-bf16 WMMA: hi*hi + hi*lo + lo*hi (drop lo*lo ~2^-16) ----
#pragma unroll
    for (int mi = 0; mi < 2; ++mi)
#pragma unroll
      for (int ni = 0; ni < 2; ++ni) {
        acc[mi][ni] = __builtin_amdgcn_wmma_f32_16x16x32_bf16(
            false, aH[mi], false, bH[ni], (short)0, acc[mi][ni], false, false);
        acc[mi][ni] = __builtin_amdgcn_wmma_f32_16x16x32_bf16(
            false, aH[mi], false, bL[ni], (short)0, acc[mi][ni], false, false);
        acc[mi][ni] = __builtin_amdgcn_wmma_f32_16x16x32_bf16(
            false, aL[mi], false, bH[ni], (short)0, acc[mi][ni], false, false);
      }
  }

  // ---- epilogue: bias + store (C layout: VGPR v -> M = v + 8*laneHalf, N = lane&15)
#pragma unroll
  for (int mi = 0; mi < 2; ++mi)
#pragma unroll
    for (int ni = 0; ni < 2; ++ni) {
      int n = n0 + wn * 32 + ni * 16 + laneRow;
      int mb = m0 + wm * 32 + mi * 16 + laneHalf * 8;
      float bias = bq[n];
#pragma unroll
      for (int v = 0; v < 8; ++v)
        q[(size_t)(mb + v) * H_DIM + n] = acc[mi][ni][v] + bias;
    }
}

// ---------------- K2: partial 4-mode DFT over s-chunks (deterministic, no atomics)
__global__ __launch_bounds__(256) void dft_partial_kernel(
    const float* __restrict__ q, const float* __restrict__ tw,
    float* __restrict__ part) {
  int b = blockIdx.x, c = blockIdx.y * 256 + threadIdx.x, ch = blockIdx.z;
  const float2* tw2 = reinterpret_cast<const float2*>(tw);
  float acc[8] = {0, 0, 0, 0, 0, 0, 0, 0};
  int s0 = ch * 128;
  for (int ss = 0; ss < 128; ++ss) {
    int s = s0 + ss;
    float qv = q[(size_t)(b * S_LEN + s) * H_DIM + c];
#pragma unroll
    for (int m = 0; m < NMODES; ++m) {
      float2 t = tw2[m * S_LEN + s];
      acc[2 * m]     += qv * t.x;   // Re: + q*cos
      acc[2 * m + 1] -= qv * t.y;   // Im: - q*sin
    }
  }
  float* dst = part + ((size_t)(b * 32 + ch) * H_DIM + c) * 8;
#pragma unroll
  for (int j = 0; j < 8; ++j) dst[j] = acc[j];
}

// ---------------- K3: reduce 32 s-chunks -> xm[b][c][m][2] ----------------------
__global__ __launch_bounds__(256) void dft_reduce_kernel(
    const float* __restrict__ part, float* __restrict__ xm) {
  int b = blockIdx.x, c = blockIdx.y * 256 + threadIdx.x;
  float acc[8] = {0, 0, 0, 0, 0, 0, 0, 0};
  for (int ch = 0; ch < 32; ++ch) {
    const float* src = part + ((size_t)(b * 32 + ch) * H_DIM + c) * 8;
#pragma unroll
    for (int j = 0; j < 8; ++j) acc[j] += src[j];
  }
  float* dst = xm + ((size_t)b * H_DIM + c) * 8;
#pragma unroll
  for (int j = 0; j < 8; ++j) dst[j] = acc[j];
}

// ---------------- K4: complex mode mixing ym[b,h,o,m] = sum_i xm * (wr + i*wi) ---
// Output stored at channel cout = o*NH + h (faithful transpose(0,3,2,1) flatten).
__global__ __launch_bounds__(64) void mode_mix_kernel(
    const float* __restrict__ xm, const float* __restrict__ wr,
    const float* __restrict__ wi, float* __restrict__ ym) {
  __shared__ float xs[DK * 8];
  int b = blockIdx.x, h = blockIdx.y, o = threadIdx.x;
  const float* src = xm + ((size_t)(b * H_DIM) + h * DK + o) * 8;  // i = o for load
#pragma unroll
  for (int j = 0; j < 8; ++j) xs[o * 8 + j] = src[j];
  __syncthreads();
  float re[4] = {0, 0, 0, 0}, im[4] = {0, 0, 0, 0};
  const float4* wr4 = reinterpret_cast<const float4*>(wr);
  const float4* wi4 = reinterpret_cast<const float4*>(wi);
  for (int i = 0; i < DK; ++i) {
    float4 a = wr4[(h * DK + i) * DK + o];
    float4 bb = wi4[(h * DK + i) * DK + o];
    float wra[4], wia[4];
    *reinterpret_cast<float4*>(wra) = a;
    *reinterpret_cast<float4*>(wia) = bb;
#pragma unroll
    for (int k = 0; k < 4; ++k) {
      float xr = xs[i * 8 + 2 * k], xi = xs[i * 8 + 2 * k + 1];
      re[k] += xr * wra[k] - xi * wia[k];
      im[k] += xr * wia[k] + xi * wra[k];
    }
  }
  int cout = o * NH + h;
  float* dst = ym + ((size_t)(b * H_DIM) + cout) * 8;
#pragma unroll
  for (int k = 0; k < 4; ++k) { dst[2 * k] = re[k]; dst[2 * k + 1] = im[k]; }
}

// ---------------- K5: 4-term inverse DFT + residual + LayerNorm ------------------
__global__ __launch_bounds__(256) void inv_ln_kernel(
    const float* __restrict__ ym, const float* __restrict__ x,
    const float* __restrict__ gamma, const float* __restrict__ beta,
    float* __restrict__ out) {
  __shared__ float yLds[H_DIM * 8];
  __shared__ float redS[8], redQ[8], bc[2];
  int b = blockIdx.x;
  int tid = threadIdx.x;
  const float* ymB = ym + (size_t)b * H_DIM * 8;
  for (int i = tid; i < H_DIM * 8; i += 256) yLds[i] = ymB[i];
  __syncthreads();
  const float invS = 1.0f / (float)S_LEN;
  for (int r = 0; r < 8; ++r) {
    int s = blockIdx.y * 8 + r;
    float cs[3], sn[3];
#pragma unroll
    for (int k = 1; k <= 3; ++k) {
      int rr = (k * s) & (S_LEN - 1);
      __sincosf((float)rr * (TWO_PI / S_LEN), &sn[k - 1], &cs[k - 1]);
    }
    float yv[4];
    float sum = 0.f, sq = 0.f;
#pragma unroll
    for (int j = 0; j < 4; ++j) {
      int c = tid + 256 * j;
      const float* Y = &yLds[c * 8];
      float v = Y[0] + 2.f * (Y[2] * cs[0] - Y[3] * sn[0] +
                              Y[4] * cs[1] - Y[5] * sn[1] +
                              Y[6] * cs[2] - Y[7] * sn[2]);
      v = v * invS + x[(size_t)(b * S_LEN + s) * H_DIM + c];
      yv[j] = v; sum += v; sq += v * v;
    }
    for (int off = 16; off > 0; off >>= 1) {
      sum += __shfl_down(sum, off);
      sq  += __shfl_down(sq, off);
    }
    int wd = tid >> 5, ln = tid & 31;
    if (ln == 0) { redS[wd] = sum; redQ[wd] = sq; }
    __syncthreads();
    if (tid == 0) {
      float ts = 0.f, tq = 0.f;
      for (int w = 0; w < 8; ++w) { ts += redS[w]; tq += redQ[w]; }
      bc[0] = ts; bc[1] = tq;
    }
    __syncthreads();
    float mu = bc[0] * (1.f / H_DIM);
    float var = bc[1] * (1.f / H_DIM) - mu * mu;
    float rstd = rsqrtf(var + 1e-5f);
#pragma unroll
    for (int j = 0; j < 4; ++j) {
      int c = tid + 256 * j;
      out[(size_t)(b * S_LEN + s) * H_DIM + c] = (yv[j] - mu) * rstd * gamma[c] + beta[c];
    }
    __syncthreads();
  }
}

extern "C" void kernel_launch(void* const* d_in, const int* in_sizes, int n_in,
                              void* d_out, int out_size, void* d_ws, size_t ws_size,
                              hipStream_t stream) {
  const float* x     = (const float*)d_in[0];
  const float* Wq    = (const float*)d_in[1];
  const float* bq    = (const float*)d_in[2];
  const float* wr    = (const float*)d_in[3];
  const float* wi    = (const float*)d_in[4];
  const float* gamma = (const float*)d_in[5];
  const float* beta  = (const float*)d_in[6];
  const int*   index = (const int*)d_in[7];
  float* out = (float*)d_out;

  float* ws   = (float*)d_ws;
  float* qws  = ws;                                        // 16384*1024        (64 MB)
  float* tw   = qws + (size_t)BATCH * S_LEN * H_DIM;       // NMODES*S*2        (128 KB)
  float* part = tw + (size_t)NMODES * S_LEN * 2;           // B*32*H*8          (4 MB)
  float* xm   = part + (size_t)BATCH * 32 * H_DIM * 8;     // B*H*8             (128 KB)
  float* ym   = xm + (size_t)BATCH * H_DIM * 8;            // B*H*8             (128 KB)

  twiddle_kernel<<<(NMODES * S_LEN) / 256, 256, 0, stream>>>(index, tw);
  qproj_kernel<<<dim3((BATCH * S_LEN) / 64, H_DIM / 128), 256, 0, stream>>>(x, Wq, bq, qws);
  dft_partial_kernel<<<dim3(BATCH, H_DIM / 256, 32), 256, 0, stream>>>(qws, tw, part);
  dft_reduce_kernel<<<dim3(BATCH, H_DIM / 256), 256, 0, stream>>>(part, xm);
  mode_mix_kernel<<<dim3(BATCH, NH), 64, 0, stream>>>(xm, wr, wi, ym);
  inv_ln_kernel<<<dim3(BATCH, S_LEN / 8), 256, 0, stream>>>(ym, x, gamma, beta, out);
}
